// CustomAttentionBlock_38405597561703
// MI455X (gfx1250) — matrix operand order repaired and checked by reading, not compiled, not run on hardware
//
#include <hip/hip_runtime.h>

// ---------------------------------------------------------------------------
// Types / helpers
// ---------------------------------------------------------------------------
typedef __attribute__((ext_vector_type(16))) __bf16 v16bf;
typedef __attribute__((ext_vector_type(8)))  float  v8f;
typedef __attribute__((ext_vector_type(4)))  unsigned uint32x4;
typedef __attribute__((ext_vector_type(8)))  int      int32x8;
typedef __attribute__((ext_vector_type(4)))  int      int32x4;

union FragU { v16bf v; unsigned u[8]; };

__device__ __forceinline__ unsigned short f2bf(float f) {
  unsigned u = __builtin_bit_cast(unsigned, f);
  u += 0x7FFFu + ((u >> 16) & 1u);             // round-to-nearest-even
  return (unsigned short)(u >> 16);
}

__device__ __forceinline__ v8f vzero() {
  v8f z = {0.f, 0.f, 0.f, 0.f, 0.f, 0.f, 0.f, 0.f};
  return z;
}

// DPP16 cross-lane move (VALU, no LDS). CTRL: 0xB1 = quad_perm(1,0,3,2) = xor1,
// 0x4E = quad_perm(2,3,0,1) = xor2, 0x141 = row_half_mirror = xor7,
// 0x140 = row_mirror = xor15. Masks {1,2,7,15} are linearly independent ->
// exact butterfly over the 16-lane half.
template <int CTRL>
__device__ __forceinline__ float dppf(float x) {
  return __builtin_bit_cast(
      float, __builtin_amdgcn_update_dpp(0, __builtin_bit_cast(int, x), CTRL,
                                         0xF, 0xF, true));
}

__device__ __forceinline__ float red_max16(float x) {
  x = fmaxf(x, dppf<0xB1>(x));
  x = fmaxf(x, dppf<0x4E>(x));
  x = fmaxf(x, dppf<0x141>(x));
  x = fmaxf(x, dppf<0x140>(x));
  return x;
}

__device__ __forceinline__ float red_sum16(float x) {
  x += dppf<0xB1>(x);
  x += dppf<0x4E>(x);
  x += dppf<0x141>(x);
  x += dppf<0x140>(x);
  return x;
}

// A fragment (16x32 bf16): lane holds row m = lane&15; vgpr j packs
// K = ((j>>2)<<4) + ((lane>>4)<<3) + 2*(j&3) .. +1   (ISA 7.12.2 table)
__device__ __forceinline__ v16bf load_frag_a(const unsigned short* base,
                                             int stride, int lane) {
  FragU f;
  const int m  = lane & 15;
  const int kh = (lane >> 4) << 3;
#pragma unroll
  for (int j = 0; j < 8; ++j) {
    int k = ((j >> 2) << 4) + kh + ((j & 3) << 1);
    f.u[j] = *(const unsigned*)(base + m * stride + k);
  }
  return f.v;
}

// B fragment (32x16 bf16): lane holds col n = lane&15; vgpr j packs
// K = ((j>>2)<<4) + ((lane>>4)<<3) + 2*(j&3) .. +1. base points at row-major
// [N][K] storage.
__device__ __forceinline__ v16bf load_frag_b(const unsigned short* base,
                                             int stride, int lane) {
  FragU f;
  const int n  = lane & 15;
  const int kh = (lane >> 4) << 3;
#pragma unroll
  for (int j = 0; j < 8; ++j)
    f.u[j] = *(const unsigned*)(base + n * stride + ((j >> 2) << 4) + kh + ((j & 3) << 1));
  return f.v;
}

__device__ __forceinline__ v8f wmma_bf16(v16bf a, v16bf b, v8f c) {
  return __builtin_amdgcn_wmma_f32_16x16x32_bf16(false, a, false, b,
                                                 (short)0, c, false, false);
}

// ---------------------------------------------------------------------------
// f32 -> bf16 weight conversion
// ---------------------------------------------------------------------------
__global__ void cvt_bf16_kernel(const float* __restrict__ in,
                                unsigned short* __restrict__ out, int n) {
  int i = blockIdx.x * 256 + threadIdx.x;
  if (i < n) out[i] = f2bf(in[i]);
}

// ---------------------------------------------------------------------------
// LayerNorm over D=768, output bf16. One row per workgroup (256 threads).
// ---------------------------------------------------------------------------
__global__ __launch_bounds__(256) void ln_kernel(
    const float* __restrict__ x, const float* __restrict__ w,
    const float* __restrict__ b, unsigned short* __restrict__ out) {
  __shared__ float red[512];
  const int row = blockIdx.x, tid = threadIdx.x;
  const float* xr = x + (size_t)row * 768;
  float v0 = xr[tid], v1 = xr[tid + 256], v2 = xr[tid + 512];
  red[tid]       = v0 + v1 + v2;
  red[tid + 256] = v0 * v0 + v1 * v1 + v2 * v2;
  __syncthreads();
  for (int off = 128; off > 0; off >>= 1) {
    if (tid < off) {
      red[tid]       += red[tid + off];
      red[tid + 256] += red[tid + 256 + off];
    }
    __syncthreads();
  }
  const float mean = red[0] * (1.f / 768.f);
  const float var  = red[256] * (1.f / 768.f) - mean * mean;
  const float rstd = rsqrtf(var + 1e-5f);
  unsigned short* orow = out + (size_t)row * 768;
  orow[tid]       = f2bf((v0 - mean) * rstd * w[tid]       + b[tid]);
  orow[tid + 256] = f2bf((v1 - mean) * rstd * w[tid + 256] + b[tid + 256]);
  orow[tid + 512] = f2bf((v2 - mean) * rstd * w[tid + 512] + b[tid + 512]);
}

// ---------------------------------------------------------------------------
// Generic bf16 WMMA GEMM:  out[M,N] = A[M,K] * Wt[N,K]^T (+bias)(+gelu)(+res)
// Block tile 128x128x32, 8 waves (2 row x 4 col), 64x32 per wave.
// ---------------------------------------------------------------------------
#define GM_BM 128
#define GM_BK 32
#define GM_LD 36   // padded LDS stride (elements); 8B aligned, conflict-free

template <bool BIAS, bool GELU, bool RES, bool OUT32>
__global__ __launch_bounds__(256) void gemm_bf16_kernel(
    const unsigned short* __restrict__ A,   // M x K bf16
    const unsigned short* __restrict__ Bw,  // N x K bf16 (nn.Linear weight)
    const float* __restrict__ bias,         // N (or null)
    const float* __restrict__ res,          // M x N f32 (or null)
    void* __restrict__ outp,                // M x N (f32 or bf16)
    int M, int N, int K) {
  __shared__ unsigned short As[2][GM_BM * GM_LD];
  __shared__ unsigned short Bs[2][GM_BM * GM_LD];

  const int tid  = threadIdx.x;
  const int lane = tid & 31;
  const int w    = tid >> 5;
  const int wr   = w >> 2;                  // 0..1
  const int wc   = w & 3;                   // 0..3
  const int rowBase = blockIdx.y * GM_BM;
  const int colBase = blockIdx.x * GM_BM;

  v8f acc[4][2];
#pragma unroll
  for (int i = 0; i < 4; ++i)
#pragma unroll
    for (int j = 0; j < 2; ++j) acc[i][j] = vzero();

  const int ldRow0 = tid >> 3;              // 0..31 (+32 per iter)
  const int ldQuad = tid & 7;               // 8 x uint2 (4 bf16) = 32 cols

  uint2 ra[4], rb[4];
#pragma unroll
  for (int i = 0; i < 4; ++i) {
    int r = ldRow0 + i * 32;
    ra[i] = *(const uint2*)(A  + (size_t)(rowBase + r) * K + ldQuad * 4);
    rb[i] = *(const uint2*)(Bw + (size_t)(colBase + r) * K + ldQuad * 4);
  }
#pragma unroll
  for (int i = 0; i < 4; ++i) {
    int r = ldRow0 + i * 32;
    *(uint2*)(&As[0][r * GM_LD + ldQuad * 4]) = ra[i];
    *(uint2*)(&Bs[0][r * GM_LD + ldQuad * 4]) = rb[i];
  }
  __syncthreads();

  const int nk = K / GM_BK;
  for (int kt = 0; kt < nk; ++kt) {
    const int cur = kt & 1;
    if (kt + 2 < nk) {
      // L2/L0 prefetch of the tile after next (global_prefetch_b8)
      const int kp = (kt + 2) * GM_BK;
      __builtin_prefetch(A  + (size_t)(rowBase + ldRow0) * K + kp, 0, 3);
      __builtin_prefetch(Bw + (size_t)(colBase + ldRow0) * K + kp, 0, 3);
    }
    if (kt + 1 < nk) {
      const int k0 = (kt + 1) * GM_BK;
#pragma unroll
      for (int i = 0; i < 4; ++i) {
        int r = ldRow0 + i * 32;
        ra[i] = *(const uint2*)(A  + (size_t)(rowBase + r) * K + k0 + ldQuad * 4);
        rb[i] = *(const uint2*)(Bw + (size_t)(colBase + r) * K + k0 + ldQuad * 4);
      }
    }
    v16bf af[4], bfv[2];
#pragma unroll
    for (int i = 0; i < 4; ++i)
      af[i] = load_frag_a(&As[cur][(wr * 64 + i * 16) * GM_LD], GM_LD, lane);
#pragma unroll
    for (int j = 0; j < 2; ++j)
      bfv[j] = load_frag_b(&Bs[cur][(wc * 32 + j * 16) * GM_LD], GM_LD, lane);
#pragma unroll
    for (int i = 0; i < 4; ++i)
#pragma unroll
      for (int j = 0; j < 2; ++j)
        acc[i][j] = wmma_bf16(af[i], bfv[j], acc[i][j]);
    __syncthreads();
    if (kt + 1 < nk) {
      const int nxt = cur ^ 1;
#pragma unroll
      for (int i = 0; i < 4; ++i) {
        int r = ldRow0 + i * 32;
        *(uint2*)(&As[nxt][r * GM_LD + ldQuad * 4]) = ra[i];
        *(uint2*)(&Bs[nxt][r * GM_LD + ldQuad * 4]) = rb[i];
      }
      __syncthreads();
    }
  }

  // Epilogue. C/D layout: row = r + 8*(lane>>4), col = lane&15 (per tile).
  const int rHalf = (lane >> 4) * 8;
  const int nLane = lane & 15;
#pragma unroll
  for (int i = 0; i < 4; ++i) {
#pragma unroll
    for (int j = 0; j < 2; ++j) {
      const int gcol = colBase + wc * 32 + j * 16 + nLane;
      const float bv = BIAS ? bias[gcol] : 0.f;
#pragma unroll
      for (int r = 0; r < 8; ++r) {
        const int grow = rowBase + wr * 64 + i * 16 + r + rHalf;
        float v = acc[i][j][r] + bv;
        if (GELU) v = 0.5f * v * (1.f + erff(v * 0.70710678118654752f));
        if (RES)  v += res[(size_t)grow * N + gcol];
        if (OUT32) ((float*)outp)[(size_t)grow * N + gcol] = v;
        else       ((unsigned short*)outp)[(size_t)grow * N + gcol] = f2bf(v);
      }
    }
  }
}

// ---------------------------------------------------------------------------
// Fused attention: one workgroup per (bc, head). T=256 keys, HD=64.
// K tile streamed into LDS by the Tensor Data Mover (2D descriptor with LDS
// padding reproducing the 72-element conflict-free stride); V transposed into
// LDS by the threads while the TDM runs; each wave owns 32 query rows; online
// (flash) softmax over 8 key-blocks of 32; P converted C->A layout through a
// per-wave LDS patch (s_wait_dscnt between store and load). Softmax row
// reductions use DPP16 butterflies (pure VALU, no LDS traffic).
// ---------------------------------------------------------------------------
#define AT_KST 72    // K LDS row stride (elems): 144B = 32 data + 4 pad DWORDs
#define AT_VST 260   // Vt LDS row stride
#define AT_PST 34    // P patch row stride
#define AT_SMEM ((256 * AT_KST + 64 * AT_VST + 8 * 32 * AT_PST) * 2)

__global__ __launch_bounds__(256) void attn_kernel(
    const unsigned short* __restrict__ qkv,  // (38912 x 2304) bf16
    unsigned short* __restrict__ out) {      // (38912 x 768) bf16
  extern __shared__ unsigned short sm[];
  unsigned short* KS = sm;                    // [256][AT_KST]  (LDS offset 0)
  unsigned short* VT = KS + 256 * AT_KST;     // [64][AT_VST] (transposed V)
  unsigned short* PP = VT + 64 * AT_VST;      // per-wave [32][AT_PST]

  const int g    = blockIdx.x;
  const int bc   = g / 12;
  const int h    = g - bc * 12;
  const int base = bc * 256;
  const int tid  = threadIdx.x;
  const int lane = tid & 31;
  const int w    = tid >> 5;

  // --- TDM: async-load the 256x64 bf16 K tile (row stride 2304 elems) into
  // KS with hardware LDS padding: pad_interval code 4 (32 DWORDs = 64 elems),
  // pad_amount code 3 (4 DWORDs = 8 elems) -> 72-element rows. Wave 0 issues.
  if (w == 0) {
    const unsigned long long ga =
        (unsigned long long)(const void*)(qkv + (size_t)base * 2304 + 768 + h * 64);
    uint32x4 g0 = {
        1u,                                   // count=1 (valid user descriptor)
        0u,                                   // lds_addr = 0 (KS at LDS base)
        (unsigned)ga,                         // global_addr[31:0]
        (unsigned)((ga >> 32) & 0x01FFFFFFu) | 0x80000000u  // addr[56:32]|type=2
    };
    int32x8 g1 = {
        (int)((1u << 16) |                    // data_size = 1 (2 bytes)
              (1u << 20) |                    // pad_enable
              (4u << 22) |                    // pad_interval: 32 DWORDs
              (3u << 25)),                    // pad_amount: 4 DWORDs
        (int)(64u << 16),                     // tensor_dim0 = 64 (bits 79:48)
        (int)(256u << 16),                    // tensor_dim1 = 256 (bits 111:80)
        (int)(64u << 16),                     // tile_dim0 = 64 (bits 127:112)
        256,                                  // tile_dim1 = 256
        2304,                                 // tensor_dim0_stride
        0, 0
    };
    int32x4 g2 = {1, 1, 0, 0};                // tensor_dim2/3 = 1 (unused)
    int32x4 g3 = {0, 0x00010000, 0, 0};       // tensor_dim4 = 1 (unused)
    int32x8 g4 = {0, 0, 0, 0, 0, 0, 0, 0};    // unused trailing group (6-arg form)
    __builtin_amdgcn_tensor_load_to_lds(g0, g1, g2, g3, g4, 0);
  }

  // --- stage V (transposed) into LDS while the TDM streams K
  {
    const unsigned short* vrow = qkv + (size_t)(base + tid) * 2304 + 1536 + h * 64;
    uint4 vd[8];
#pragma unroll
    for (int q = 0; q < 8; ++q) vd[q] = ((const uint4*)vrow)[q];
    union { uint4 u4; unsigned short s[8]; } t;
#pragma unroll
    for (int q = 0; q < 8; ++q) {
      t.u4 = vd[q];
#pragma unroll
      for (int e = 0; e < 8; ++e) VT[(q * 8 + e) * AT_VST + tid] = t.s[e];
    }
  }
  if (w == 0) __builtin_amdgcn_s_wait_tensorcnt(0);
  __syncthreads();

  // --- Q fragments for this wave's 32 rows (2 row-tiles x 2 K-slices of 32)
  const int tok0 = base + w * 32;
  v16bf qf[2][2];
#pragma unroll
  for (int rt = 0; rt < 2; ++rt)
#pragma unroll
    for (int kt = 0; kt < 2; ++kt)
      qf[rt][kt] = load_frag_a(
          qkv + (size_t)(tok0 + rt * 16) * 2304 + h * 64 + kt * 32, 2304, lane);

  float Mst[2][8], Lst[2][8];
  v8f O[2][4];
#pragma unroll
  for (int rt = 0; rt < 2; ++rt) {
#pragma unroll
    for (int r = 0; r < 8; ++r) { Mst[rt][r] = -1e30f; Lst[rt][r] = 0.f; }
#pragma unroll
    for (int c = 0; c < 4; ++c) O[rt][c] = vzero();
  }

  unsigned short* myP = PP + w * 32 * AT_PST;
  const float SCALE = 0.125f;   // 64^-0.5

  for (int kb = 0; kb < 8; ++kb) {
    // rt-invariant B fragments for this key block: hoisted so each is loaded
    // once per kb (the dscnt fence below would otherwise block CSE).
    v16bf kf[2][2];   // [ctile][kt]
#pragma unroll
    for (int kt = 0; kt < 2; ++kt) {
      kf[0][kt] = load_frag_b(&KS[(kb * 32 +  0) * AT_KST + kt * 32], AT_KST, lane);
      kf[1][kt] = load_frag_b(&KS[(kb * 32 + 16) * AT_KST + kt * 32], AT_KST, lane);
    }
    v16bf vf[4];
#pragma unroll
    for (int c = 0; c < 4; ++c)
      vf[c] = load_frag_b(&VT[(c * 16) * AT_VST + kb * 32], AT_VST, lane);

#pragma unroll
    for (int rt = 0; rt < 2; ++rt) {
      // S tiles for keys [kb*32, kb*32+32): two 16x16 C-frags, K=64 chained.
      v8f s0 = vzero(), s1 = vzero();
#pragma unroll
      for (int kt = 0; kt < 2; ++kt) {
        s0 = wmma_bf16(qf[rt][kt], kf[0][kt], s0);
        s1 = wmma_bf16(qf[rt][kt], kf[1][kt], s1);
      }
      // online softmax over this 32-key block (DPP butterflies, no LDS)
      float p0[8], p1[8], al[8];
#pragma unroll
      for (int r = 0; r < 8; ++r) {
        float a = s0[r] * SCALE, b = s1[r] * SCALE;
        float tm = red_max16(fmaxf(a, b));
        float nm = fmaxf(Mst[rt][r], tm);
        al[r] = __expf(Mst[rt][r] - nm);
        Mst[rt][r] = nm;
        p0[r] = __expf(a - nm);
        p1[r] = __expf(b - nm);
        float rs = red_sum16(p0[r] + p1[r]);
        Lst[rt][r] = Lst[rt][r] * al[r] + rs;
      }
#pragma unroll
      for (int c = 0; c < 4; ++c)
#pragma unroll
        for (int r = 0; r < 8; ++r) O[rt][c][r] *= al[r];

      // P: C-layout -> bf16 patch -> A-layout fragment
      const int prow0 = rt * 16 + (lane >> 4) * 8;
#pragma unroll
      for (int r = 0; r < 8; ++r) {
        unsigned short* pr = myP + (prow0 + r) * AT_PST;
        pr[lane & 15]        = f2bf(p0[r]);
        pr[16 + (lane & 15)] = f2bf(p1[r]);
      }
      asm volatile("s_wait_dscnt 0" ::: "memory");
      v16bf pa = load_frag_a(myP + rt * 16 * AT_PST, AT_PST, lane);
#pragma unroll
      for (int c = 0; c < 4; ++c)
        O[rt][c] = wmma_bf16(pa, vf[c], O[rt][c]);
    }
  }

  // --- normalize and store (bf16)
  const int rHalf = (lane >> 4) * 8;
  const int nLane = lane & 15;
#pragma unroll
  for (int rt = 0; rt < 2; ++rt)
#pragma unroll
    for (int c = 0; c < 4; ++c)
#pragma unroll
      for (int r = 0; r < 8; ++r) {
        int row = tok0 + rt * 16 + r + rHalf;
        int col = h * 64 + c * 16 + nLane;
        out[(size_t)row * 768 + col] = f2bf(O[rt][c][r] / Lst[rt][r]);
      }
}

// ---------------------------------------------------------------------------
// Host side
// ---------------------------------------------------------------------------
extern "C" void kernel_launch(void* const* d_in, const int* in_sizes, int n_in,
                              void* d_out, int out_size, void* d_ws,
                              size_t ws_size, hipStream_t stream) {
  (void)in_sizes; (void)n_in; (void)out_size; (void)ws_size;
  const int M = 38912;            // B * C * T tokens
  const int D = 768, HID = 3072, QKVN = 2304;

  const float* x       = (const float*)d_in[0];
  const float* norm1_w = (const float*)d_in[1];
  const float* norm1_b = (const float*)d_in[2];
  const float* qkv_w   = (const float*)d_in[3];
  const float* proj_w  = (const float*)d_in[4];
  const float* proj_b  = (const float*)d_in[5];
  const float* norm2_w = (const float*)d_in[6];
  const float* norm2_b = (const float*)d_in[7];
  const float* fc1_w   = (const float*)d_in[8];
  const float* fc1_b   = (const float*)d_in[9];
  const float* fc2_w   = (const float*)d_in[10];
  const float* fc2_b   = (const float*)d_in[11];
  float* outp = (float*)d_out;

  char* ws = (char*)d_ws;
  size_t off = 0;
  auto alloc = [&](size_t bytes) {
    void* p = ws + off;
    off = (off + bytes + 255) & ~(size_t)255;
    return p;
  };
  unsigned short* wqkv  = (unsigned short*)alloc((size_t)QKVN * D * 2);
  unsigned short* wproj = (unsigned short*)alloc((size_t)D * D * 2);
  unsigned short* wfc1  = (unsigned short*)alloc((size_t)HID * D * 2);
  unsigned short* wfc2  = (unsigned short*)alloc((size_t)D * HID * 2);
  unsigned short* lnbuf = (unsigned short*)alloc((size_t)M * D * 2);      // ln1 / ln2
  unsigned short* qkvb  = (unsigned short*)alloc((size_t)M * QKVN * 2);
  unsigned short* attnb = (unsigned short*)alloc((size_t)M * D * 2);
  float*          xres  = (float*)alloc((size_t)M * D * 4);
  unsigned short* h1    = (unsigned short*)alloc((size_t)M * HID * 2);

  // weights -> bf16
  {
    int n;
    n = QKVN * D; cvt_bf16_kernel<<<(n + 255) / 256, 256, 0, stream>>>(qkv_w,  wqkv,  n);
    n = D * D;    cvt_bf16_kernel<<<(n + 255) / 256, 256, 0, stream>>>(proj_w, wproj, n);
    n = HID * D;  cvt_bf16_kernel<<<(n + 255) / 256, 256, 0, stream>>>(fc1_w,  wfc1,  n);
    n = D * HID;  cvt_bf16_kernel<<<(n + 255) / 256, 256, 0, stream>>>(fc2_w,  wfc2,  n);
  }

  // LN1 -> bf16
  ln_kernel<<<M, 256, 0, stream>>>(x, norm1_w, norm1_b, lnbuf);

  // QKV GEMM: (M x 768) x (2304 x 768)^T -> bf16
  gemm_bf16_kernel<false, false, false, false>
      <<<dim3(QKVN / 128, M / 128), 256, 0, stream>>>(
          lnbuf, wqkv, nullptr, nullptr, qkvb, M, QKVN, D);

  // fused attention (152 * 12 head-batches)
  attn_kernel<<<152 * 12, 256, AT_SMEM, stream>>>(qkvb, attnb);

  // proj GEMM + bias + residual(x) -> f32 xres
  gemm_bf16_kernel<true, false, true, true>
      <<<dim3(D / 128, M / 128), 256, 0, stream>>>(
          attnb, wproj, proj_b, x, xres, M, D, D);

  // LN2 -> bf16 (reuse lnbuf)
  ln_kernel<<<M, 256, 0, stream>>>(xres, norm2_w, norm2_b, lnbuf);

  // FC1 + bias + exact GELU -> bf16
  gemm_bf16_kernel<true, true, false, false>
      <<<dim3(HID / 128, M / 128), 256, 0, stream>>>(
          lnbuf, wfc1, fc1_b, nullptr, h1, M, HID, D);

  // FC2 + bias + residual(xres) -> f32 output
  gemm_bf16_kernel<true, false, true, true>
      <<<dim3(D / 128, M / 128), 256, 0, stream>>>(
          h1, wfc2, fc2_b, xres, outp, M, D, HID);
}
